// SpatialGCN_91130616087328
// MI455X (gfx1250) — matrix operand order repaired and checked by reading, not compiled, not run on hardware
//
#include <hip/hip_runtime.h>
#include <hip/hip_bf16.h>
#include <stdint.h>

#define C    2048
#define P    16
#define NF   512          // B*T frames
#define EPS  1e-5f

typedef __attribute__((ext_vector_type(16))) __bf16          v16bf;
typedef __attribute__((ext_vector_type(8)))  float           v8f;
typedef __attribute__((ext_vector_type(8)))  unsigned short  ushort8;

union Frag { v16bf bf; ushort8 u[2]; };

__device__ __forceinline__ unsigned short f2bf(float f) {
    union { float f; uint32_t u; } v; v.f = f;
    uint32_t r = v.u + 0x7FFFu + ((v.u >> 16) & 1u);   // round-to-nearest-even
    return (unsigned short)(r >> 16);
}

// ---------------- weight fp32 -> bf16 ----------------
__global__ void cvt_bf16_kernel(const float* __restrict__ src,
                                unsigned short* __restrict__ dst, int n) {
    int i = blockIdx.x * blockDim.x + threadIdx.x;
    if (i < n) dst[i] = f2bf(src[i]);
}

// ---------------- per-frame mean over P -> bf16 ----------------
__global__ void mean_kernel(const float* __restrict__ X,
                            unsigned short* __restrict__ Mb) {
    int i = blockIdx.x * blockDim.x + threadIdx.x;     // over NF*C
    if (i >= NF * C) return;
    int f = i / C, c = i % C;
    const float* x = X + ((size_t)f * P) * C + c;
    float s = 0.f;
#pragma unroll
    for (int p = 0; p < P; ++p) s += x[(size_t)p * C];
    Mb[i] = f2bf(s * (1.0f / P));
}

// ---------------- attention GEMM + sigmoid ----------------
// attn[n,d] = sigmoid( sum_c Mean[n,c]*W_att[d,c] + b_att[d] )
__global__ void attn_gemm_kernel(const unsigned short* __restrict__ Mb,
                                 const unsigned short* __restrict__ Wab,
                                 const float* __restrict__ b_att,
                                 float* __restrict__ attn) {
    int wave = blockIdx.x * (blockDim.x >> 5) + (threadIdx.x >> 5);
    int lane = threadIdx.x & 31;
    int rowTile = wave >> 7;          // 32 row tiles of 16 frames
    int colTile = wave & 127;         // 128 col tiles of 16
    if (rowTile >= NF / 16) return;

    int half = lane >> 4, lr = lane & 15;
    const unsigned short* aptr = Mb + ((size_t)(rowTile * 16 + lr)) * C + half * 8;
    const unsigned short* bptr = Wab + ((size_t)(colTile * 16 + lr)) * C + half * 8;

    v8f acc = {};
    for (int k0 = 0; k0 < C; k0 += 32) {
        Frag a, b;
        a.u[0] = *(const ushort8*)(aptr + k0);
        a.u[1] = *(const ushort8*)(aptr + k0 + 16);
        b.u[0] = *(const ushort8*)(bptr + k0);
        b.u[1] = *(const ushort8*)(bptr + k0 + 16);
        acc = __builtin_amdgcn_wmma_f32_16x16x32_bf16(false, a.bf, false, b.bf,
                                                      (short)0, acc, false, false);
    }
    int col = colTile * 16 + lr;
    int rbase = rowTile * 16 + half * 8;
    float bb = b_att[col];
#pragma unroll
    for (int v = 0; v < 8; ++v) {
        float r = acc[v] + bb;
        attn[(size_t)(rbase + v) * C + col] = 1.0f / (1.0f + __expf(-r));
    }
}

// ---------------- gate X by attn, emit bf16 ----------------
__global__ void gate_kernel(const float* __restrict__ X,
                            const float* __restrict__ attn,
                            unsigned short* __restrict__ Xgb) {
    size_t i = (size_t)blockIdx.x * blockDim.x + threadIdx.x;
    if (i >= (size_t)NF * P * C) return;
    int c = (int)(i % C);
    int f = (int)(i / ((size_t)P * C));
    Xgb[i] = f2bf(X[i] * attn[(size_t)f * C + c]);
}

// ---------------- GEMM1 + fused per-frame BN + ReLU ----------------
// Each wave: TWO frames (2 x 16 rows) x 64 output channels (2x4 WMMA tiles).
// B-fragments (weights) are reused across both frames: 12 loads / 8 WMMA.
// BN over P is wave-local per frame: column n lives in lanes {n, n+16}.
__global__ void gemm1_bn_kernel(const unsigned short* __restrict__ Xgb,
                                const unsigned short* __restrict__ W1b,
                                const float* __restrict__ b1,
                                const float* __restrict__ gamma,
                                const float* __restrict__ beta,
                                unsigned short* __restrict__ Hb) {
    int wave = blockIdx.x * (blockDim.x >> 5) + (threadIdx.x >> 5);
    int lane = threadIdx.x & 31;
    int fp = wave >> 5;               // frame pair: 0..255
    int cg = wave & 31;               // 32 column-groups of 64
    if (fp >= NF / 2) return;
    int f0 = fp * 2;
    int d0 = cg * 64;
    int half = lane >> 4, lr = lane & 15;

    const unsigned short* ap0 = Xgb + ((size_t)f0 * P + lr) * C + half * 8;
    const unsigned short* ap1 = ap0 + (size_t)P * C;
    const unsigned short* bp[4];
#pragma unroll
    for (int t = 0; t < 4; ++t)
        bp[t] = W1b + ((size_t)(d0 + t * 16 + lr)) * C + half * 8;

    v8f acc[2][4] = {};
    for (int k0 = 0; k0 < C; k0 += 32) {
        Frag a0, a1;
        a0.u[0] = *(const ushort8*)(ap0 + k0);
        a0.u[1] = *(const ushort8*)(ap0 + k0 + 16);
        a1.u[0] = *(const ushort8*)(ap1 + k0);
        a1.u[1] = *(const ushort8*)(ap1 + k0 + 16);
#pragma unroll
        for (int t = 0; t < 4; ++t) {
            Frag b;
            b.u[0] = *(const ushort8*)(bp[t] + k0);
            b.u[1] = *(const ushort8*)(bp[t] + k0 + 16);
            acc[0][t] = __builtin_amdgcn_wmma_f32_16x16x32_bf16(false, a0.bf, false, b.bf,
                                                                (short)0, acc[0][t], false, false);
            acc[1][t] = __builtin_amdgcn_wmma_f32_16x16x32_bf16(false, a1.bf, false, b.bf,
                                                                (short)0, acc[1][t], false, false);
        }
    }

    int rbase = half * 8;
#pragma unroll
    for (int fi = 0; fi < 2; ++fi) {
        int frame = f0 + fi;
#pragma unroll
        for (int t = 0; t < 4; ++t) {
            int col = d0 + t * 16 + lr;
            float bb = b1[col];
            float h[8], s1 = 0.f, s2 = 0.f;
#pragma unroll
            for (int v = 0; v < 8; ++v) {
                h[v] = acc[fi][t][v] + bb;
                s1 += h[v];
                s2 += h[v] * h[v];
            }
            s1 += __shfl_xor(s1, 16, 32);     // combine the two half-columns
            s2 += __shfl_xor(s2, 16, 32);
            float mu  = s1 * (1.0f / 16.0f);
            float var = s2 * (1.0f / 16.0f) - mu * mu;
            float g   = gamma[col] * rsqrtf(var + EPS);
            float be  = beta[col];
#pragma unroll
            for (int v = 0; v < 8; ++v) {
                float hn = (h[v] - mu) * g + be;
                hn = fmaxf(hn, 0.0f);
                Hb[(size_t)(frame * 16 + rbase + v) * C + col] = f2bf(hn);
            }
        }
    }
}

// ---------------- GEMM2 + bias -> fp32 output ----------------
// Same 2-frame x 4-tile blocking (no BN constraint, but same reuse win).
__global__ void gemm2_kernel(const unsigned short* __restrict__ Hb,
                             const unsigned short* __restrict__ W2b,
                             const float* __restrict__ b2,
                             float* __restrict__ out) {
    int wave = blockIdx.x * (blockDim.x >> 5) + (threadIdx.x >> 5);
    int lane = threadIdx.x & 31;
    int fp = wave >> 5;
    int cg = wave & 31;
    if (fp >= NF / 2) return;
    int f0 = fp * 2;
    int d0 = cg * 64;
    int half = lane >> 4, lr = lane & 15;

    const unsigned short* ap0 = Hb + ((size_t)f0 * P + lr) * C + half * 8;
    const unsigned short* ap1 = ap0 + (size_t)P * C;
    const unsigned short* bp[4];
#pragma unroll
    for (int t = 0; t < 4; ++t)
        bp[t] = W2b + ((size_t)(d0 + t * 16 + lr)) * C + half * 8;

    v8f acc[2][4] = {};
    for (int k0 = 0; k0 < C; k0 += 32) {
        Frag a0, a1;
        a0.u[0] = *(const ushort8*)(ap0 + k0);
        a0.u[1] = *(const ushort8*)(ap0 + k0 + 16);
        a1.u[0] = *(const ushort8*)(ap1 + k0);
        a1.u[1] = *(const ushort8*)(ap1 + k0 + 16);
#pragma unroll
        for (int t = 0; t < 4; ++t) {
            Frag b;
            b.u[0] = *(const ushort8*)(bp[t] + k0);
            b.u[1] = *(const ushort8*)(bp[t] + k0 + 16);
            acc[0][t] = __builtin_amdgcn_wmma_f32_16x16x32_bf16(false, a0.bf, false, b.bf,
                                                                (short)0, acc[0][t], false, false);
            acc[1][t] = __builtin_amdgcn_wmma_f32_16x16x32_bf16(false, a1.bf, false, b.bf,
                                                                (short)0, acc[1][t], false, false);
        }
    }

    int rbase = half * 8;
#pragma unroll
    for (int fi = 0; fi < 2; ++fi) {
        int frame = f0 + fi;
#pragma unroll
        for (int t = 0; t < 4; ++t) {
            int col = d0 + t * 16 + lr;
            float bb = b2[col];
#pragma unroll
            for (int v = 0; v < 8; ++v)
                out[(size_t)(frame * 16 + rbase + v) * C + col] = acc[fi][t][v] + bb;
        }
    }
}

extern "C" void kernel_launch(void* const* d_in, const int* in_sizes, int n_in,
                              void* d_out, int out_size, void* d_ws, size_t ws_size,
                              hipStream_t stream) {
    (void)in_sizes; (void)n_in; (void)out_size; (void)ws_size;

    const float* X     = (const float*)d_in[0];
    const float* W_att = (const float*)d_in[1];
    const float* b_att = (const float*)d_in[2];
    const float* W1    = (const float*)d_in[3];
    const float* b1    = (const float*)d_in[4];
    const float* gamma = (const float*)d_in[5];
    const float* beta  = (const float*)d_in[6];
    const float* W2    = (const float*)d_in[7];
    const float* b2    = (const float*)d_in[8];
    float* out = (float*)d_out;

    char* ws = (char*)d_ws;
    unsigned short* Wab = (unsigned short*)(ws);                       //  8 MB
    unsigned short* W1b = (unsigned short*)(ws + ((size_t)8  << 20));  //  8 MB
    unsigned short* W2b = (unsigned short*)(ws + ((size_t)16 << 20));  //  8 MB
    unsigned short* Mb  = (unsigned short*)(ws + ((size_t)24 << 20));  //  2 MB
    float*          At  = (float*)        (ws + ((size_t)28 << 20));   //  4 MB
    unsigned short* Xgb = (unsigned short*)(ws + ((size_t)36 << 20));  // 32 MB
    unsigned short* Hb  = (unsigned short*)(ws + ((size_t)72 << 20));  // 32 MB

    const int nW = C * C;                     // 4M elements per weight
    cvt_bf16_kernel<<<nW / 256, 256, 0, stream>>>(W_att, Wab, nW);
    cvt_bf16_kernel<<<nW / 256, 256, 0, stream>>>(W1,    W1b, nW);
    cvt_bf16_kernel<<<nW / 256, 256, 0, stream>>>(W2,    W2b, nW);

    mean_kernel<<<(NF * C) / 256, 256, 0, stream>>>(X, Mb);

    // (NF/16)*(C/16) = 4096 waves, 8 waves/block
    attn_gemm_kernel<<<512, 256, 0, stream>>>(Mb, Wab, b_att, At);

    gate_kernel<<<(NF * P * C) / 256, 256, 0, stream>>>(X, At, Xgb);

    // (NF/2) * (C/64) = 8192 waves, 8 waves/block
    gemm1_bn_kernel<<<1024, 256, 0, stream>>>(Xgb, W1b, b1, gamma, beta, Hb);
    gemm2_kernel  <<<1024, 256, 0, stream>>>(Hb, W2b, b2, out);
}
